// Cotracker_4621384810703
// MI455X (gfx1250) — matrix operand order repaired and checked by reading, not compiled, not run on hardware
//
#include <hip/hip_runtime.h>
#include <hip/hip_bf16.h>
#include <math.h>

// ---------------------------------------------------------------------------
// CoTracker-style transformer on MI455X (gfx1250, wave32, WMMA bf16 16x16x32)
// - v = src @ val_w hoisted out of the 6-iter refine loop (reference
//   recomputes it per _msda call -> 6x redundant 44 GFLOP GEMM + 1.4GB traffic)
// - pyramid srcs stored as bf16 (only consumer is the bf16 WMMA GEMM):
//   saves ~200MB HBM traffic on a bandwidth-bound pipeline (23.3 TB/s)
// - GEMM: 64x128 block tile, 2x4 v_wmma per wave, global_prefetch of next tile
// ---------------------------------------------------------------------------

typedef __bf16 bf16_t;
typedef __attribute__((ext_vector_type(16))) __bf16 v16bf;
typedef __attribute__((ext_vector_type(8)))  float  v8f;

#define L_ALL 66836

// ---------------------------------------------------------------------------
// WMMA GEMM: out[M,N] = act(A[M,K] @ W[N,K]^T + bias[N]), optional relu.
// Block = 128 threads = 4 waves (2x2); block tile 64x128; each wave 32x64
// (2x4 of v_wmma_f32_16x16x32_bf16). f32 A/W converted to bf16 in registers.
// A frag (16x32): lane&15 = row; lane<16 -> K {0..7,16..23} else {8..15,24..31}.
// B frag (32x16): lane&15 = col; lane<16 -> K {0..7,16..23} else {8..15,24..31}.
// C/D: vgpr r -> row r + (lane>=16)*8, col = lane&15.
// ---------------------------------------------------------------------------
__global__ __launch_bounds__(128)
void k_gemm(const float* __restrict__ A, const float* __restrict__ W,
            const float* __restrict__ bias, float* __restrict__ out,
            int M, int N, int K, int relu)
{
  int wave = threadIdx.x >> 5;
  int lane = threadIdx.x & 31;
  int wm = wave >> 1, wn = wave & 1;
  int m0 = blockIdx.x * 64 + wm * 32;
  int n0 = blockIdx.y * 128 + wn * 64;
  if (m0 >= M || n0 >= N) return;   // wave-uniform -> EXEC stays all-ones

  int lhalf = lane >> 4;
  int l15   = lane & 15;

  v8f acc[2][4];
#pragma unroll
  for (int i = 0; i < 2; i++)
#pragma unroll
    for (int j = 0; j < 4; j++) acc[i][j] = (v8f){0.f};

  for (int kb = 0; kb < K; kb += 32) {
    if (kb + 32 < K) {   // gfx1250 global_prefetch_b8 of the next K tile
      __builtin_prefetch(A + (size_t)(m0 + l15) * K + kb + 32, 0, 1);
      __builtin_prefetch(W + (size_t)(n0 + l15) * K + kb + 32, 0, 1);
    }
    v16bf a[2], b[4];
#pragma unroll
    for (int i = 0; i < 2; i++) {
      int row = m0 + i * 16 + l15; if (row >= M) row = M - 1;
      const float* ap = A + (size_t)row * K + kb + lhalf * 8;
#pragma unroll
      for (int t = 0; t < 8; t++) a[i][t]     = (bf16_t)ap[t];
#pragma unroll
      for (int t = 0; t < 8; t++) a[i][8 + t] = (bf16_t)ap[16 + t];
    }
#pragma unroll
    for (int j = 0; j < 4; j++) {
      int col = n0 + j * 16 + l15; if (col >= N) col = N - 1;
      const float* wp = W + (size_t)col * K + kb + lhalf * 8;
#pragma unroll
      for (int t = 0; t < 8; t++) b[j][t]     = (bf16_t)wp[t];
#pragma unroll
      for (int t = 0; t < 8; t++) b[j][8 + t] = (bf16_t)wp[16 + t];
    }
#pragma unroll
    for (int i = 0; i < 2; i++)
#pragma unroll
      for (int j = 0; j < 4; j++)
        acc[i][j] = __builtin_amdgcn_wmma_f32_16x16x32_bf16(
            false, a[i], false, b[j], (short)0, acc[i][j], false, false);
  }

#pragma unroll
  for (int i = 0; i < 2; i++)
#pragma unroll
    for (int j = 0; j < 4; j++) {
      int col = n0 + j * 16 + l15;
      if (col >= N) continue;
      float bv = bias ? bias[col] : 0.f;
#pragma unroll
      for (int r = 0; r < 8; r++) {
        int row = m0 + i * 16 + r + lhalf * 8;
        if (row >= M) continue;
        float v = acc[i][j][r] + bv;
        if (relu) v = fmaxf(v, 0.f);
        out[(size_t)row * N + col] = v;
      }
    }
}

// Same GEMM but A is already bf16 (pyramid srcs): halves A-side HBM traffic.
__global__ __launch_bounds__(128)
void k_gemm_bA(const bf16_t* __restrict__ A, const float* __restrict__ W,
               const float* __restrict__ bias, float* __restrict__ out,
               int M, int N, int K, int relu)
{
  int wave = threadIdx.x >> 5;
  int lane = threadIdx.x & 31;
  int wm = wave >> 1, wn = wave & 1;
  int m0 = blockIdx.x * 64 + wm * 32;
  int n0 = blockIdx.y * 128 + wn * 64;
  if (m0 >= M || n0 >= N) return;

  int lhalf = lane >> 4;
  int l15   = lane & 15;

  v8f acc[2][4];
#pragma unroll
  for (int i = 0; i < 2; i++)
#pragma unroll
    for (int j = 0; j < 4; j++) acc[i][j] = (v8f){0.f};

  for (int kb = 0; kb < K; kb += 32) {
    if (kb + 32 < K) {
      __builtin_prefetch(A + (size_t)(m0 + l15) * K + kb + 32, 0, 1);
      __builtin_prefetch(W + (size_t)(n0 + l15) * K + kb + 32, 0, 1);
    }
    v16bf a[2], b[4];
#pragma unroll
    for (int i = 0; i < 2; i++) {
      int row = m0 + i * 16 + l15; if (row >= M) row = M - 1;
      const bf16_t* ap = A + (size_t)row * K + kb + lhalf * 8;
#pragma unroll
      for (int t = 0; t < 8; t++) a[i][t]     = ap[t];
#pragma unroll
      for (int t = 0; t < 8; t++) a[i][8 + t] = ap[16 + t];
    }
#pragma unroll
    for (int j = 0; j < 4; j++) {
      int col = n0 + j * 16 + l15; if (col >= N) col = N - 1;
      const float* wp = W + (size_t)col * K + kb + lhalf * 8;
#pragma unroll
      for (int t = 0; t < 8; t++) b[j][t]     = (bf16_t)wp[t];
#pragma unroll
      for (int t = 0; t < 8; t++) b[j][8 + t] = (bf16_t)wp[16 + t];
    }
#pragma unroll
    for (int i = 0; i < 2; i++)
#pragma unroll
      for (int j = 0; j < 4; j++)
        acc[i][j] = __builtin_amdgcn_wmma_f32_16x16x32_bf16(
            false, a[i], false, b[j], (short)0, acc[i][j], false, false);
  }

#pragma unroll
  for (int i = 0; i < 2; i++)
#pragma unroll
    for (int j = 0; j < 4; j++) {
      int col = n0 + j * 16 + l15;
      if (col >= N) continue;
      float bv = bias ? bias[col] : 0.f;
#pragma unroll
      for (int r = 0; r < 8; r++) {
        int row = m0 + i * 16 + r + lhalf * 8;
        if (row >= M) continue;
        float v = acc[i][j][r] + bv;
        if (relu) v = fmaxf(v, 0.f);
        out[(size_t)row * N + col] = v;
      }
    }
}

// ---------------------------------------------------------------------------
// Attention: one wave per (b, head, query). grid.x = B*8*Lq.
// ---------------------------------------------------------------------------
__global__ __launch_bounds__(32)
void k_attn(const float* __restrict__ Q, const float* __restrict__ Kp,
            const float* __restrict__ Vp, float* __restrict__ O,
            int Lq, int Lk, int qstride, int kstride)
{
  __shared__ float sc[1024];
  __shared__ float qs[32];
  int t = blockIdx.x;
  int q = t % Lq; t /= Lq;
  int h = t & 7;  int b = t >> 3;
  int lane = threadIdx.x;

  const float* qp = Q + ((size_t)(b * Lq + q)) * qstride + h * 32;
  qs[lane] = qp[lane];
  __syncthreads();

  float mx = -1e30f;
  for (int k = lane; k < Lk; k += 32) {
    const float* kp = Kp + ((size_t)(b * Lk + k)) * kstride + h * 32;
    float s = 0.f;
#pragma unroll
    for (int d = 0; d < 32; d++) s += qs[d] * kp[d];
    s *= 0.17677669529663687f;   // 1/sqrt(32)
    sc[k] = s;
    mx = fmaxf(mx, s);
  }
#pragma unroll
  for (int o = 16; o > 0; o >>= 1) mx = fmaxf(mx, __shfl_xor(mx, o));
  float sum = 0.f;
  for (int k = lane; k < Lk; k += 32) {
    float e = __expf(sc[k] - mx); sc[k] = e; sum += e;
  }
#pragma unroll
  for (int o = 16; o > 0; o >>= 1) sum += __shfl_xor(sum, o);
  __syncthreads();
  float inv = 1.f / sum;

  float acc = 0.f;
  for (int k = 0; k < Lk; k++)
    acc += sc[k] * Vp[((size_t)(b * Lk + k)) * kstride + h * 32 + lane];
  O[((size_t)(b * Lq + q)) * 256 + h * 32 + lane] = acc * inv;
}

// ---------------------------------------------------------------------------
// Residual + LayerNorm(256) + optional PE add. One block per row.
// out = LN(a + b) * g + beta (+ pe)
// ---------------------------------------------------------------------------
__global__ __launch_bounds__(256)
void k_add_ln(const float* __restrict__ a, const float* __restrict__ b,
              const float* __restrict__ g, const float* __restrict__ beta,
              const float* __restrict__ pe, float* __restrict__ out)
{
  __shared__ float rs[8], rq[8];
  int row = blockIdx.x, c = threadIdx.x;
  size_t idx = (size_t)row * 256 + c;
  float x = a[idx] + (b ? b[idx] : 0.f);
  float s1 = x, s2 = x * x;
#pragma unroll
  for (int o = 16; o > 0; o >>= 1) { s1 += __shfl_xor(s1, o); s2 += __shfl_xor(s2, o); }
  int lane = c & 31, wv = c >> 5;
  if (lane == 0) { rs[wv] = s1; rq[wv] = s2; }
  __syncthreads();
  if (c == 0) {
    float t1 = 0.f, t2 = 0.f;
#pragma unroll
    for (int i = 0; i < 8; i++) { t1 += rs[i]; t2 += rq[i]; }
    rs[0] = t1; rq[0] = t2;
  }
  __syncthreads();
  float mean = rs[0] * (1.f / 256.f);
  float var  = rq[0] * (1.f / 256.f) - mean * mean;
  float y = (x - mean) * rsqrtf(var + 1e-5f) * g[c] + beta[c];
  if (pe) y += pe[idx];
  out[idx] = y;
}

// ---------------------------------------------------------------------------
// Pyramid build: antialiased resize (half-pixel triangle filter like
// jax.image.resize) + LayerNorm(src_ln) + sine pos2d + level embedding.
// One block per (image, token); thread = channel. Output stored bf16 (only
// consumer is the bf16 WMMA value-projection GEMM).
// ---------------------------------------------------------------------------
__global__ __launch_bounds__(256)
void k_build_src(const float* __restrict__ feats, bf16_t* __restrict__ src,
                 const float* __restrict__ ln_g, const float* __restrict__ ln_b,
                 const float* __restrict__ level_pos)
{
  __shared__ float rs[8], rq[8];
  int blk = blockIdx.x;
  int b = blk / L_ALL;
  int l = blk % L_ALL;
  int lvl, s, start;
  if      (l <   196) { lvl = 0; s =  14; start =     0; }
  else if (l <   980) { lvl = 1; s =  28; start =   196; }
  else if (l <  4116) { lvl = 2; s =  56; start =   980; }
  else if (l < 16660) { lvl = 3; s = 112; start =  4116; }
  else                { lvl = 4; s = 224; start = 16660; }
  int loc = l - start;
  int y = loc / s, x = loc % s;
  int c = threadIdx.x;
  const float* f = feats + ((size_t)b * 256 + c) * (224 * 224);

  float rv;
  if (s == 224) {
    rv = f[y * 224 + x];
  } else {
    float scale = (float)s / 224.f;
    float inv   = 224.f / (float)s;
    float cy = ((float)y + 0.5f) * inv - 0.5f;
    float cx = ((float)x + 0.5f) * inv - 0.5f;
    int y0 = (int)ceilf(cy - inv);  if (y0 < 0)   y0 = 0;
    int y1 = (int)floorf(cy + inv); if (y1 > 223) y1 = 223;
    int x0 = (int)ceilf(cx - inv);  if (x0 < 0)   x0 = 0;
    int x1 = (int)floorf(cx + inv); if (x1 > 223) x1 = 223;
    float num = 0.f, wys = 0.f, wxs = 0.f;
    for (int jx = x0; jx <= x1; jx++) {
      float wx = 1.f - fabsf((float)jx - cx) * scale;
      if (wx > 0.f) wxs += wx;
    }
    for (int jy = y0; jy <= y1; jy++) {
      float wy = 1.f - fabsf((float)jy - cy) * scale;
      if (wy <= 0.f) continue;
      wys += wy;
      float rowacc = 0.f;
      const float* fr = f + jy * 224;
      for (int jx = x0; jx <= x1; jx++) {
        float wx = 1.f - fabsf((float)jx - cx) * scale;
        if (wx > 0.f) rowacc += wx * fr[jx];
      }
      num += wy * rowacc;
    }
    rv = num / (wys * wxs);
  }

  // block LayerNorm over 256 channels
  float s1 = rv, s2 = rv * rv;
#pragma unroll
  for (int o = 16; o > 0; o >>= 1) { s1 += __shfl_xor(s1, o); s2 += __shfl_xor(s2, o); }
  int lane = c & 31, wv = c >> 5;
  if (lane == 0) { rs[wv] = s1; rq[wv] = s2; }
  __syncthreads();
  if (c == 0) {
    float t1 = 0.f, t2 = 0.f;
#pragma unroll
    for (int i = 0; i < 8; i++) { t1 += rs[i]; t2 += rq[i]; }
    rs[0] = t1; rq[0] = t2;
  }
  __syncthreads();
  float mean = rs[0] * (1.f / 256.f);
  float var  = rq[0] * (1.f / 256.f) - mean * mean;
  float yv = (rv - mean) * rsqrtf(var + 1e-5f) * ln_g[c] + ln_b[c];

  // DETR sine pos2d + level embedding
  int cc = c & 127;
  int isx = (c >= 128);
  float e = ((isx ? (float)(x + 1) : (float)(y + 1)) / ((float)s + 1e-6f))
            * 6.283185307179586f;
  int m = cc >> 1;
  float dim_t = __expf(9.210340371976184f * ((float)m * (1.f / 64.f))); // 10000^(2m/128)
  float pr = e / dim_t;
  float pos = (cc & 1) ? __cosf(pr) : __sinf(pr);
  pos += level_pos[lvl * 256 + c];

  src[((size_t)b * L_ALL + l) * 256 + c] = (bf16_t)(yv + pos);
}

// ---------------------------------------------------------------------------
// Multi-scale deformable attention sampling. One block per (frame, query);
// thread = (head h = t>>5, dim d = t&31). v: [NF, 66836, 256].
// ---------------------------------------------------------------------------
__global__ __launch_bounds__(256)
void k_msda(const float* __restrict__ off, const float* __restrict__ awl,
            const float* __restrict__ pts, const float* __restrict__ v,
            float* __restrict__ out)
{
  int r = blockIdx.x;
  int b = r >> 8;                // 256 queries per frame
  int t = threadIdx.x;
  int h = t >> 5, d = t & 31;
  float rx = pts[r * 2 + 0], ry = pts[r * 2 + 1];

  const float* al = awl + (size_t)r * 320 + h * 40;
  float mx = -1e30f;
#pragma unroll
  for (int i = 0; i < 40; i++) mx = fmaxf(mx, al[i]);
  float aw[40]; float ssum = 0.f;
#pragma unroll
  for (int i = 0; i < 40; i++) { aw[i] = __expf(al[i] - mx); ssum += aw[i]; }
  float inv = 1.f / ssum;

  const float* op = off + (size_t)r * 640 + h * 80;
  const int sizes[5] = {14, 28, 56, 112, 224};
  float acc = 0.f;
  int start = 0;
#pragma unroll
  for (int lvl = 0; lvl < 5; lvl++) {
    int s = sizes[lvl];
    const float* vb = v + ((size_t)b * L_ALL + start) * 256 + h * 32 + d;
#pragma unroll
    for (int p = 0; p < 8; p++) {
      float ox = op[(lvl * 8 + p) * 2 + 0];
      float oy = op[(lvl * 8 + p) * 2 + 1];
      float lx = rx + ox / (float)s;
      float ly = ry + oy / (float)s;
      float xf = lx * s - 0.5f, yf = ly * s - 0.5f;
      float xfl = floorf(xf), yfl = floorf(yf);
      int x0 = (int)xfl, y0 = (int)yfl;
      float wx = xf - xfl, wy = yf - yfl;
      float sv = 0.f;
      { int xi = x0,     yi = y0;
        if (xi >= 0 && xi < s && yi >= 0 && yi < s)
          sv += (1.f - wx) * (1.f - wy) * vb[(size_t)(yi * s + xi) * 256]; }
      { int xi = x0 + 1, yi = y0;
        if (xi >= 0 && xi < s && yi >= 0 && yi < s)
          sv += wx * (1.f - wy) * vb[(size_t)(yi * s + xi) * 256]; }
      { int xi = x0,     yi = y0 + 1;
        if (xi >= 0 && xi < s && yi >= 0 && yi < s)
          sv += (1.f - wx) * wy * vb[(size_t)(yi * s + xi) * 256]; }
      { int xi = x0 + 1, yi = y0 + 1;
        if (xi >= 0 && xi < s && yi >= 0 && yi < s)
          sv += wx * wy * vb[(size_t)(yi * s + xi) * 256]; }
      acc += aw[lvl * 8 + p] * inv * sv;
    }
    start += s * s;
  }
  out[(size_t)r * 256 + t] = acc;
}

// ---------------------------------------------------------------------------
// small helpers
// ---------------------------------------------------------------------------
__global__ void k_pe1d(float* __restrict__ out)
{
  int pos = blockIdx.x, c = threadIdx.x;
  int i2 = (c >> 1) << 1;
  float div = __expf((float)i2 * (-9.210340371976184f / 256.f));
  float a = (float)pos * div;
  out[(size_t)pos * 256 + c] = (c & 1) ? __cosf(a) : __sinf(a);
}

__global__ void k_pts(const float* __restrict__ p, float* __restrict__ pf,
                      float* __restrict__ po)
{
  int i = blockIdx.x * 256 + threadIdx.x;
  if (i < 512)  pf[i] = p[i] * (1.f / 224.f);
  if (i < 2048) po[i] = p[512 + i] * (1.f / 224.f);
}

__global__ void k_bcast(const float* __restrict__ src, float* __restrict__ dst,
                        int n, int period)
{
  int i = blockIdx.x * 256 + threadIdx.x;
  if (i < n) dst[i] = src[i % period];
}

__global__ void k_add(const float* __restrict__ a, const float* __restrict__ b,
                      float* __restrict__ out, int n)
{
  int i = blockIdx.x * 256 + threadIdx.x;
  if (i < n) out[i] = a[i] + b[i];
}

__global__ void k_dxy(const float* __restrict__ mlp, float* __restrict__ pts,
                      float* __restrict__ outp)
{
  int i = blockIdx.x * 256 + threadIdx.x;
  if (i < 2048) {
    float d = (1.f / (1.f + __expf(-mlp[i])) - 0.5f) * 0.2f;
    float np = pts[i] + d;
    pts[i] = np;
    outp[i] = np * 224.f;
  }
}

// ---------------------------------------------------------------------------
// host orchestration
// ---------------------------------------------------------------------------
enum {
  IN_FEATS_FIR = 0, IN_FEATS_OTH, IN_POINTS,
  P_LEVEL_POS, P_QUERY_EMBED,
  P_SRC_LN_G, P_SRC_LN_B, P_DL_LN1_G, P_DL_LN1_B, P_DL_LN2_G, P_DL_LN2_B,
  P_DL_LN3_G, P_DL_LN3_B, P_FIR_LN_G, P_FIR_LN_B, P_ENC_LN1_G, P_ENC_LN1_B,
  P_ENC_LN2_G, P_ENC_LN2_B, P_OTH_LN_G, P_OTH_LN_B, P_DE_LN1_G, P_DE_LN1_B,
  P_DE_LN2_G, P_DE_LN2_B, P_DE_LN3_G, P_DE_LN3_B,
  P_SA_IN_W, P_SA_IN_B, P_SA_OUT_W, P_SA_OUT_B,
  P_OFF_W, P_OFF_B, P_AW_W, P_AW_B, P_VAL_W, P_VAL_B, P_MO_W, P_MO_B,
  P_DL_FF1_W, P_DL_FF1_B, P_DL_FF2_W, P_DL_FF2_B,
  P_ENC_IN_W, P_ENC_IN_B, P_ENC_OUT_W, P_ENC_OUT_B,
  P_ENC_FF1_W, P_ENC_FF1_B, P_ENC_FF2_W, P_ENC_FF2_B,
  P_DE_SA_IN_W, P_DE_SA_IN_B, P_DE_SA_OUT_W, P_DE_SA_OUT_B,
  P_DE_CA_IN_W, P_DE_CA_IN_B, P_DE_CA_OUT_W, P_DE_CA_OUT_B,
  P_DE_FF1_W, P_DE_FF1_B, P_DE_FF2_W, P_DE_FF2_B,
  P_DXY_W1, P_DXY_B1, P_DXY_W2, P_DXY_B2, P_DXY_W3, P_DXY_B3,
  P_DQ_W1, P_DQ_B1, P_DQ_W2, P_DQ_B2, P_DQ_W3, P_DQ_B3
};

static inline void gemm(hipStream_t st, const float* A, const float* W,
                        const float* bias, float* out, int M, int N, int K,
                        int relu)
{
  dim3 g((M + 63) / 64, (N + 127) / 128);
  k_gemm<<<g, 128, 0, st>>>(A, W, bias, out, M, N, K, relu);
}

static inline void gemmB(hipStream_t st, const bf16_t* A, const float* W,
                         const float* bias, float* out, int M, int N, int K,
                         int relu)
{
  dim3 g((M + 63) / 64, (N + 127) / 128);
  k_gemm_bA<<<g, 128, 0, st>>>(A, W, bias, out, M, N, K, relu);
}

extern "C" void kernel_launch(void* const* d_in, const int* in_sizes, int n_in,
                              void* d_out, int out_size, void* d_ws, size_t ws_size,
                              hipStream_t stream)
{
  auto in = [&](int i) { return (const float*)d_in[i]; };
  float* outp = (float*)d_out;

  // ---- workspace layout (byte allocator, 256B aligned) ----
  char* base = (char*)d_ws;
  size_t off = 0;
  auto allocB = [&](size_t bytes) {
    void* p = base + off; off += (bytes + 255) & ~(size_t)255; return p;
  };
  auto allocF = [&](size_t n) { return (float*)allocB(n * sizeof(float)); };

  bf16_t* SRCB  = (bf16_t*)allocB((size_t)4 * L_ALL * 256 * sizeof(bf16_t));
  float* VFIR   = allocF((size_t)L_ALL * 256);
  float* VOTH   = allocF((size_t)4 * L_ALL * 256);
  float* PE_P   = allocF(256 * 256);
  float* PE_FP  = allocF(1024 * 256);
  float* S0     = allocF(1024 * 1024);              // ff hidden
  float* S1     = allocF(1024 * 768);               // qkv
  float* S2     = allocF(1024 * 256);
  float* S3     = allocF(1024 * 256);
  float* S5     = allocF(1024 * 640);               // offsets
  float* S6     = allocF(1024 * 320);               // attn-weight logits
  float* T1     = allocF(1024 * 256);
  float* T2     = allocF(1024 * 256);
  float* T3     = allocF(1024 * 256);
  float* FEATSQ = allocF(1024 * 256);
  float* X1     = allocF(1024 * 256);
  float* XA     = allocF(1024 * 256);
  float* XB     = allocF(1024 * 256);
  float* XOUT   = allocF(1024 * 256);
  float* QC     = allocF(1024 * 256);
  float* KBUF   = allocF(256 * 256);
  float* VBUF   = allocF(256 * 256);
  float* QBUF   = allocF(256 * 256);
  float* FLN    = allocF(256 * 256);
  float* E0     = allocF(256 * 256);
  float* FIRMEM = allocF(256 * 256);
  float* QUERIES= allocF(1024 * 256);
  float* DQ     = allocF(1024 * 256);
  float* PTS_F  = allocF(512);
  float* PTS_O  = allocF(2048);
  float* DXY    = allocF(2048);
  (void)ws_size; (void)n_in; (void)in_sizes; (void)out_size;

  // ---- setup ----
  k_pe1d<<<256, 256, 0, stream>>>(PE_P);
  k_pe1d<<<1024, 256, 0, stream>>>(PE_FP);
  k_pts<<<8, 256, 0, stream>>>(in(IN_POINTS), PTS_F, PTS_O);
  k_bcast<<<256, 256, 0, stream>>>(in(P_QUERY_EMBED), QBUF, 65536, 65536);
  k_bcast<<<1024, 256, 0, stream>>>(in(P_QUERY_EMBED), QUERIES, 262144, 65536);

  // pyramid + value projections (computed ONCE)
  k_build_src<<<L_ALL, 256, 0, stream>>>(in(IN_FEATS_FIR), SRCB,
      in(P_SRC_LN_G), in(P_SRC_LN_B), in(P_LEVEL_POS));
  gemmB(stream, SRCB, in(P_VAL_W), in(P_VAL_B), VFIR, L_ALL, 256, 256, 0);
  k_build_src<<<4 * L_ALL, 256, 0, stream>>>(in(IN_FEATS_OTH), SRCB,
      in(P_SRC_LN_G), in(P_SRC_LN_B), in(P_LEVEL_POS));
  gemmB(stream, SRCB, in(P_VAL_W), in(P_VAL_B), VOTH, 4 * L_ALL, 256, 256, 0);

  // ---- deformable decoder layer (shared weights), R rows, NF frames ----
  auto def_dec_layer = [&](const float* tgt0, const float* pts, const float* v,
                           int R, int NF, float* result) {
    gemm(stream, tgt0, in(P_SA_IN_W), in(P_SA_IN_B), S1, R, 768, 256, 0);
    k_attn<<<NF * 8 * 256, 32, 0, stream>>>(S1, S1 + 256, S1 + 512, S2,
                                            256, 256, 768, 768);
    gemm(stream, S2, in(P_SA_OUT_W), in(P_SA_OUT_B), S3, R, 256, 256, 0);
    k_add_ln<<<R, 256, 0, stream>>>(tgt0, S3, in(P_DL_LN2_G), in(P_DL_LN2_B),
                                    nullptr, T1);
    gemm(stream, T1, in(P_OFF_W), in(P_OFF_B), S5, R, 640, 256, 0);
    gemm(stream, T1, in(P_AW_W),  in(P_AW_B),  S6, R, 320, 256, 0);
    k_msda<<<R, 256, 0, stream>>>(S5, S6, pts, v, S2);
    gemm(stream, S2, in(P_MO_W), in(P_MO_B), S3, R, 256, 256, 0);
    k_add_ln<<<R, 256, 0, stream>>>(T1, S3, in(P_DL_LN1_G), in(P_DL_LN1_B),
                                    nullptr, T2);
    gemm(stream, T2, in(P_DL_FF1_W), in(P_DL_FF1_B), S0, R, 1024, 256, 1);
    gemm(stream, S0, in(P_DL_FF2_W), in(P_DL_FF2_B), S3, R, 256, 1024, 0);
    k_add_ln<<<R, 256, 0, stream>>>(T2, S3, in(P_DL_LN3_G), in(P_DL_LN3_B),
                                    nullptr, result);
  };

  // ---- first frame -> memory ----
  def_dec_layer(QBUF, PTS_F, VFIR, 256, 1, T3);
  k_add_ln<<<256, 256, 0, stream>>>(T3, nullptr, in(P_FIR_LN_G), in(P_FIR_LN_B),
                                    nullptr, FLN);
  k_add<<<256, 256, 0, stream>>>(FLN, PE_P, E0, 65536);
  gemm(stream, E0, in(P_ENC_IN_W), in(P_ENC_IN_B), S1, 256, 768, 256, 0);
  k_attn<<<8 * 256, 32, 0, stream>>>(S1, S1 + 256, S1 + 512, S2, 256, 256, 768, 768);
  gemm(stream, S2, in(P_ENC_OUT_W), in(P_ENC_OUT_B), S3, 256, 256, 256, 0);
  k_add_ln<<<256, 256, 0, stream>>>(E0, S3, in(P_ENC_LN1_G), in(P_ENC_LN1_B),
                                    nullptr, T1);
  gemm(stream, T1, in(P_ENC_FF1_W), in(P_ENC_FF1_B), S0, 256, 1024, 256, 1);
  gemm(stream, S0, in(P_ENC_FF2_W), in(P_ENC_FF2_B), S3, 256, 256, 1024, 0);
  k_add_ln<<<256, 256, 0, stream>>>(T1, S3, in(P_ENC_LN2_G), in(P_ENC_LN2_B),
                                    nullptr, FIRMEM);

  // ---- other frames: 6 refinement iterations ----
  for (int it = 0; it < 6; it++) {
    def_dec_layer(QUERIES, PTS_O, VOTH, 1024, 4, FEATSQ);
    k_add_ln<<<1024, 256, 0, stream>>>(FEATSQ, nullptr, in(P_OTH_LN_G),
                                       in(P_OTH_LN_B), PE_FP, X1);
    // decoder layer: self-attn over 1024 tokens
    gemm(stream, X1, in(P_DE_SA_IN_W), in(P_DE_SA_IN_B), S1, 1024, 768, 256, 0);
    k_attn<<<8 * 1024, 32, 0, stream>>>(S1, S1 + 256, S1 + 512, S2,
                                        1024, 1024, 768, 768);
    gemm(stream, S2, in(P_DE_SA_OUT_W), in(P_DE_SA_OUT_B), S3, 1024, 256, 256, 0);
    k_add_ln<<<1024, 256, 0, stream>>>(X1, S3, in(P_DE_LN1_G), in(P_DE_LN1_B),
                                       nullptr, XA);
    // cross-attn with fir_mem (256 tokens)
    gemm(stream, XA, in(P_DE_CA_IN_W), in(P_DE_CA_IN_B), QC, 1024, 256, 256, 0);
    gemm(stream, FIRMEM, in(P_DE_CA_IN_W) + 256 * 256, in(P_DE_CA_IN_B) + 256,
         KBUF, 256, 256, 256, 0);
    gemm(stream, FIRMEM, in(P_DE_CA_IN_W) + 512 * 256, in(P_DE_CA_IN_B) + 512,
         VBUF, 256, 256, 256, 0);
    k_attn<<<8 * 1024, 32, 0, stream>>>(QC, KBUF, VBUF, S2, 1024, 256, 256, 256);
    gemm(stream, S2, in(P_DE_CA_OUT_W), in(P_DE_CA_OUT_B), S3, 1024, 256, 256, 0);
    k_add_ln<<<1024, 256, 0, stream>>>(XA, S3, in(P_DE_LN2_G), in(P_DE_LN2_B),
                                       nullptr, XB);
    gemm(stream, XB, in(P_DE_FF1_W), in(P_DE_FF1_B), S0, 1024, 1024, 256, 1);
    gemm(stream, S0, in(P_DE_FF2_W), in(P_DE_FF2_B), S3, 1024, 256, 1024, 0);
    k_add_ln<<<1024, 256, 0, stream>>>(XB, S3, in(P_DE_LN3_G), in(P_DE_LN3_B),
                                       nullptr, XOUT);
    // dxy head: sigmoid MLP -> point update + coord output
    gemm(stream, XOUT, in(P_DXY_W1), in(P_DXY_B1), S2, 1024, 256, 256, 1);
    gemm(stream, S2,   in(P_DXY_W2), in(P_DXY_B2), S3, 1024, 256, 256, 1);
    gemm(stream, S3,   in(P_DXY_W3), in(P_DXY_B3), DXY, 1024, 2, 256, 0);
    k_dxy<<<8, 256, 0, stream>>>(DXY, PTS_O, outp + (size_t)it * 2048);
    // query update
    gemm(stream, FEATSQ, in(P_DQ_W1), in(P_DQ_B1), S2, 1024, 256, 256, 1);
    gemm(stream, S2,     in(P_DQ_W2), in(P_DQ_B2), S3, 1024, 256, 256, 1);
    gemm(stream, S3,     in(P_DQ_W3), in(P_DQ_B3), DQ, 1024, 256, 256, 0);
    k_add<<<1024, 256, 0, stream>>>(QUERIES, DQ, QUERIES, 262144);
  }
}
